// QuantLinear_61314953118436
// MI455X (gfx1250) — 8-bit matrix operand order repaired and checked by reading, compiled, not run on hardware
//
#include <hip/hip_runtime.h>
#include <hip/hip_bf16.h>

typedef int v8i __attribute__((ext_vector_type(8)));

typedef int b128_t __attribute__((vector_size(16)));
typedef __attribute__((address_space(1))) b128_t* gptr128_t;
typedef __attribute__((address_space(3))) b128_t* lptr128_t;

#define QMAXF 127.0f
#define EPSF 1e-12f

#if __has_builtin(__builtin_amdgcn_global_load_async_to_lds_b128) && \
    __has_builtin(__builtin_amdgcn_s_wait_asynccnt)
#define USE_ASYNC_LDS 1
#else
#define USE_ASYNC_LDS 0
#endif

__global__ void zero_u32(unsigned int* __restrict__ p, int n) {
    int i = blockIdx.x * blockDim.x + threadIdx.x;
    if (i < n) p[i] = 0u;
}

__global__ void colmax_kernel(const float* __restrict__ src, int rows, int K,
                              unsigned int* __restrict__ outmax) {
    int row0 = blockIdx.x * 128;
    int rend = row0 + 128;
    if (rend > rows) rend = rows;
    for (int c = threadIdx.x; c < K; c += blockDim.x) {
        float m = 0.0f;
        for (int r = row0; r < rend; ++r)
            m = fmaxf(m, fabsf(src[(size_t)r * K + c]));
        atomicMax(&outmax[c], __float_as_uint(m));
    }
}

__global__ void scale_kernel(const unsigned int* __restrict__ inmax,
                             const unsigned int* __restrict__ wmax,
                             float* __restrict__ smooth,
                             float* __restrict__ invs, int K) {
    int k = blockIdx.x * blockDim.x + threadIdx.x;
    if (k >= K) return;
    float a = __uint_as_float(inmax[k]);
    float b = __uint_as_float(wmax[k]);
    float s = sqrtf(fmaxf(a * b, EPSF));
    smooth[k] = s;
    invs[k]   = 1.0f / s;
}

__global__ void quant_rows(const float* __restrict__ src,
                           const float* __restrict__ colscale, int K,
                           signed char* __restrict__ q,
                           float* __restrict__ rowscale, int nrows) {
    const int wave = threadIdx.x >> 5;
    const int lane = threadIdx.x & 31;
    const int row  = blockIdx.x * (blockDim.x >> 5) + wave;
    if (row >= nrows) return;

    const int iters = K / 128;
    float4 vals[8];
    float m = 0.0f;
    const float4* s4 = (const float4*)(src + (size_t)row * K);
    const float4* c4 = (const float4*)colscale;
    for (int i = 0; i < iters; ++i) {
        int g = lane + 32 * i;
        float4 v = s4[g];
        float4 cs = c4[g];
        v.x *= cs.x; v.y *= cs.y; v.z *= cs.z; v.w *= cs.w;
        vals[i] = v;
        m = fmaxf(m, fmaxf(fmaxf(fabsf(v.x), fabsf(v.y)),
                           fmaxf(fabsf(v.z), fabsf(v.w))));
    }
    for (int off = 16; off >= 1; off >>= 1)
        m = fmaxf(m, __shfl_xor(m, off, 32));
    float rs = fmaxf(m * (1.0f / QMAXF), EPSF);
    if (lane == 0) rowscale[row] = rs;
    float inv = 1.0f / rs;

    unsigned int* q32 = (unsigned int*)(q + (size_t)row * K);
    for (int i = 0; i < iters; ++i) {
        int g = lane + 32 * i;
        float4 v = vals[i];
        int a = (int)fminf(fmaxf(rintf(v.x * inv), -QMAXF), QMAXF);
        int b = (int)fminf(fmaxf(rintf(v.y * inv), -QMAXF), QMAXF);
        int c = (int)fminf(fmaxf(rintf(v.z * inv), -QMAXF), QMAXF);
        int d = (int)fminf(fmaxf(rintf(v.w * inv), -QMAXF), QMAXF);
        q32[g] = (unsigned)(a & 0xff) | ((unsigned)(b & 0xff) << 8) |
                 ((unsigned)(c & 0xff) << 16) | ((unsigned)(d & 0xff) << 24);
    }
}

#define BROWP 80

__global__ __launch_bounds__(256, 2)
void gemm_iu8_dequant(const signed char* __restrict__ xq,
                      const signed char* __restrict__ wq,
                      const float* __restrict__ xs,
                      const float* __restrict__ wsc,
                      const float* __restrict__ bias,
                      float* __restrict__ out,
                      int M, int N, int K) {
    __shared__ __align__(16) unsigned char btile[64 * BROWP];

    const int t     = threadIdx.x;
    const int wave  = t >> 5;
    const int lane  = t & 31;
    const int half  = lane >> 4;
    const int lrow  = lane & 15;
    const int tileN = blockIdx.x * 64;
    const int tileM = blockIdx.y * 256 + wave * 32;
    if (tileM >= M) return;

    v8i acc[2][4] = {};

    const size_t arow0 = (size_t)(tileM + lrow) * (size_t)K;
    const size_t arow1 = (size_t)(tileM + 16 + lrow) * (size_t)K;
    const int aoff0 = half ? 16 : 0;

    const int bn = t >> 2;
    const int bk = (t & 3) * 16;

    for (int k0 = 0; k0 < K; k0 += 64) {
        {
            const signed char* gsrc = wq + (size_t)(tileN + bn) * K + (k0 + bk);
            unsigned char* ldst = &btile[bn * BROWP + bk];
#if USE_ASYNC_LDS
            __builtin_amdgcn_global_load_async_to_lds_b128(
                (gptr128_t)(gsrc), (lptr128_t)(ldst), 0, 0);
            __builtin_amdgcn_s_wait_asynccnt(0);
#else
            *(uint4*)ldst = *(const uint4*)gsrc;
#endif
        }
        __syncthreads();

        const signed char* ap0 = xq + arow0 + k0;
        const signed char* ap1 = xq + arow1 + k0;
        v8i a0, a1;
        {
            uint2 c0 = *(const uint2*)(ap0 + aoff0);
            uint2 c1 = *(const uint2*)(ap0 + aoff0 + 8);
            uint2 c2 = *(const uint2*)(ap0 + aoff0 + 32);
            uint2 c3 = *(const uint2*)(ap0 + aoff0 + 40);
            a0[0] = (int)c0.x; a0[1] = (int)c0.y;
            a0[2] = (int)c1.x; a0[3] = (int)c1.y;
            a0[4] = (int)c2.x; a0[5] = (int)c2.y;
            a0[6] = (int)c3.x; a0[7] = (int)c3.y;
        }
        {
            uint2 c0 = *(const uint2*)(ap1 + aoff0);
            uint2 c1 = *(const uint2*)(ap1 + aoff0 + 8);
            uint2 c2 = *(const uint2*)(ap1 + aoff0 + 32);
            uint2 c3 = *(const uint2*)(ap1 + aoff0 + 40);
            a1[0] = (int)c0.x; a1[1] = (int)c0.y;
            a1[2] = (int)c1.x; a1[3] = (int)c1.y;
            a1[4] = (int)c2.x; a1[5] = (int)c2.y;
            a1[6] = (int)c3.x; a1[7] = (int)c3.y;
        }
        if (k0 + 64 < K) {
            __builtin_prefetch(ap0 + 64, 0, 3);
            __builtin_prefetch(ap1 + 64, 0, 3);
        }

#pragma unroll
        for (int nt = 0; nt < 4; ++nt) {
            const unsigned char* bp =
                &btile[(nt * 16 + lrow) * BROWP + half * 16];
            uint4 b0 = *(const uint4*)(bp);
            uint4 b1 = *(const uint4*)(bp + 32);
            v8i b;
            b[0] = (int)b0.x; b[1] = (int)b0.y; b[2] = (int)b0.z; b[3] = (int)b0.w;
            b[4] = (int)b1.x; b[5] = (int)b1.y; b[6] = (int)b1.z; b[7] = (int)b1.w;
            acc[0][nt] = __builtin_amdgcn_wmma_i32_16x16x64_iu8(
                true, a0, true, b, acc[0][nt], false, false);
            acc[1][nt] = __builtin_amdgcn_wmma_i32_16x16x64_iu8(
                true, a1, true, b, acc[1][nt], false, false);
        }
        __syncthreads();
    }

#pragma unroll
    for (int ms = 0; ms < 2; ++ms) {
#pragma unroll
        for (int nt = 0; nt < 4; ++nt) {
            const int n = tileN + nt * 16 + lrow;
            const float wn = wsc[n];
            const float bn_ = bias[n];
#pragma unroll
            for (int g = 0; g < 8; ++g) {
                const int m = tileM + ms * 16 + half * 8 + g;
                float v = (float)acc[ms][nt][g] * xs[m] * wn + bn_;
                __builtin_nontemporal_store(v, &out[(size_t)m * N + n]);
            }
        }
    }
}

extern "C" void kernel_launch(void* const* d_in, const int* in_sizes, int n_in,
                              void* d_out, int out_size, void* d_ws, size_t ws_size,
                              hipStream_t stream) {
    const float* x    = (const float*)d_in[0];
    const float* w    = (const float*)d_in[1];
    const float* bias = (const float*)d_in[2];
    float* out = (float*)d_out;

    const int N = in_sizes[2];
    const int K = in_sizes[1] / N;
    const int M = in_sizes[0] / K;

    char* ws = (char*)d_ws;
    size_t off = 0;
    auto take = [&](size_t bytes) -> char* {
        char* p = ws + off;
        off = (off + bytes + 255) & ~(size_t)255;
        return p;
    };
    signed char* xq   = (signed char*)take((size_t)M * K);
    signed char* wq   = (signed char*)take((size_t)N * K);
    float* xs         = (float*)take((size_t)M * 4);
    float* wsc        = (float*)take((size_t)N * 4);
    unsigned* inmax   = (unsigned*)take((size_t)K * 4);
    unsigned* wmax    = (unsigned*)take((size_t)K * 4);
    float* smooth     = (float*)take((size_t)K * 4);
    float* invs       = (float*)take((size_t)K * 4);
    (void)ws_size; (void)n_in; (void)out_size;

    zero_u32<<<(K + 255) / 256, 256, 0, stream>>>(inmax, K);
    zero_u32<<<(K + 255) / 256, 256, 0, stream>>>(wmax, K);

    colmax_kernel<<<(M + 127) / 128, 256, 0, stream>>>(x, M, K, inmax);
    colmax_kernel<<<(N + 127) / 128, 256, 0, stream>>>(w, N, K, wmax);

    scale_kernel<<<(K + 255) / 256, 256, 0, stream>>>(inmax, wmax, smooth, invs, K);

    quant_rows<<<(M + 7) / 8, 256, 0, stream>>>(x, invs, K, xq, xs, M);
    quant_rows<<<(N + 7) / 8, 256, 0, stream>>>(w, smooth, K, wq, wsc, N);

    dim3 grid(N / 64, (M + 255) / 256);
    gemm_iu8_dequant<<<grid, 256, 0, stream>>>(xq, wq, xs, wsc, bias, out, M, N, K);
}
